// SelfAttention_46428596470281
// MI455X (gfx1250) — matrix operand order repaired and checked by reading, not compiled, not run on hardware
//
#include <hip/hip_runtime.h>

#define D_  1024
#define L_  2048
#define H_  16
#define HD_ 64

typedef __attribute__((ext_vector_type(16))) __bf16 v16bf;
typedef __attribute__((ext_vector_type(8)))  __bf16 v8bf;
typedef __attribute__((ext_vector_type(4)))  __bf16 v4bf;
typedef __attribute__((ext_vector_type(8)))  float  v8f;

static __device__ __forceinline__ __bf16 f2bf(float f) {
  unsigned u = __builtin_bit_cast(unsigned, f);
  u += 0x7FFFu + ((u >> 16) & 1u);                       // round-to-nearest-even
  unsigned short s = (unsigned short)(u >> 16);
  return __builtin_bit_cast(__bf16, s);
}

static __device__ __forceinline__ v16bf cat8(v8bf lo, v8bf hi) {
  return __builtin_shufflevector(lo, hi, 0,1,2,3,4,5,6,7,8,9,10,11,12,13,14,15);
}

static __device__ __forceinline__ v8f wmma_bf16(v16bf a, v16bf b, v8f c) {
  return __builtin_amdgcn_wmma_f32_16x16x32_bf16(false, a, false, b, (short)0, c,
                                                 false, false);
}

static __device__ __forceinline__ v8f zero8() {
  v8f z = {0.f,0.f,0.f,0.f,0.f,0.f,0.f,0.f};
  return z;
}

// CDNA5 async copy: global -> LDS, 16B per lane, tracked by ASYNCcnt.
// lds_off = low 32 bits of a generic pointer to __shared__ (flat->LDS truncates).
static __device__ __forceinline__ void async_ld128(unsigned lds_off,
                                                   const __bf16* g) {
  asm volatile("global_load_async_to_lds_b128 %0, %1, off"
               :: "v"(lds_off), "v"(g) : "memory");
}
#define WAIT_ASYNC_LE4() asm volatile("s_wait_asynccnt 0x4" ::: "memory")
#define WAIT_ASYNC_0()   asm volatile("s_wait_asynccnt 0x0" ::: "memory")

// ---------------- f32 -> bf16 conversion (vectorized) ----------------
__global__ void cvt_kernel(const float* __restrict__ src, __bf16* __restrict__ dst,
                           int n4) {
  int i = blockIdx.x * blockDim.x + threadIdx.x;
  if (i >= n4) return;
  float4 v = reinterpret_cast<const float4*>(src)[i];
  v4bf r = { f2bf(v.x), f2bf(v.y), f2bf(v.z), f2bf(v.w) };
  reinterpret_cast<v4bf*>(dst)[i] = r;
}

// ---------------- fused QKV projection: out = x @ W^T + b ----------------
// grid: (M/128, N/128, 3), block 256 (8 waves, 2x4 wave grid)
__global__ __launch_bounds__(256) void qkv_kernel(
    const __bf16* __restrict__ xh,
    const __bf16* __restrict__ wq, const __bf16* __restrict__ wk,
    const __bf16* __restrict__ wv,
    const float* __restrict__ bq, const float* __restrict__ bk,
    const float* __restrict__ bv,
    __bf16* __restrict__ qo, __bf16* __restrict__ ko, __bf16* __restrict__ vto) {
  __shared__ __bf16 sA[2][128 * 40];   // padded stride: 80B rows, conflict-free
  __shared__ __bf16 sB[2][128 * 40];

  const int z = blockIdx.z;
  const __bf16* w   = (z == 0) ? wq : (z == 1) ? wk : wv;
  const float* bias = (z == 0) ? bq : (z == 1) ? bk : bv;
  const int mb = blockIdx.x * 128;
  const int nb = blockIdx.y * 128;
  const int tid = threadIdx.x;
  const int lane = tid & 31;
  const int wid = tid >> 5;
  const int wm = (wid & 1) * 64;
  const int wn = (wid >> 1) * 32;
  const int half16 = lane & 15;
  const int ak  = (lane < 16) ? 0 : 8;    // A chunk: K {0-7,16-23} | {8-15,24-31}
  const int bk_ = (lane < 16) ? 0 : 8;    // B chunk: K {0-7,16-23} | {8-15,24-31}
  const int lr = tid >> 2;                // cooperative-load row 0..63
  const int lc = (tid & 3) * 8;           // 16B chunk within 32-wide k tile

  v8f acc[4][2];
#pragma unroll
  for (int mi = 0; mi < 4; ++mi) { acc[mi][0] = zero8(); acc[mi][1] = zero8(); }

  const __bf16* ga0 = &xh[(size_t)(mb + lr) * D_ + lc];
  const __bf16* ga1 = &xh[(size_t)(mb + lr + 64) * D_ + lc];
  const __bf16* gb0 = &w[(size_t)(nb + lr) * D_ + lc];
  const __bf16* gb1 = &w[(size_t)(nb + lr + 64) * D_ + lc];
  const unsigned la0 = (unsigned)(size_t)&sA[0][lr * 40 + lc];
  const unsigned la1 = (unsigned)(size_t)&sA[0][(lr + 64) * 40 + lc];
  const unsigned lb0 = (unsigned)(size_t)&sB[0][lr * 40 + lc];
  const unsigned lb1 = (unsigned)(size_t)&sB[0][(lr + 64) * 40 + lc];
  const unsigned bstepB = 128 * 40 * 2;   // bytes between double buffers
  const unsigned bstepE = 128 * 40;       // elements between double buffers

  auto issue = [&](int k0, unsigned o) {   // o: uniform 0 or bstepB
    async_ld128(la0 + o, ga0 + k0);
    async_ld128(la1 + o, ga1 + k0);
    async_ld128(lb0 + o, gb0 + k0);
    async_ld128(lb1 + o, gb1 + k0);
  };

  auto compute = [&](const __bf16* A, const __bf16* B) {
    v16bf bfrag[2];
#pragma unroll
    for (int ni = 0; ni < 2; ++ni) {
      int col = wn + ni * 16 + half16;
      bfrag[ni] = cat8(*reinterpret_cast<const v8bf*>(&B[col * 40 + bk_]),
                       *reinterpret_cast<const v8bf*>(&B[col * 40 + bk_ + 16]));
    }
#pragma unroll
    for (int mi = 0; mi < 4; ++mi) {
      int row = wm + mi * 16 + half16;
      v16bf afrag = cat8(*reinterpret_cast<const v8bf*>(&A[row * 40 + ak]),
                         *reinterpret_cast<const v8bf*>(&A[row * 40 + ak + 16]));
      acc[mi][0] = wmma_bf16(afrag, bfrag[0], acc[mi][0]);
      acc[mi][1] = wmma_bf16(afrag, bfrag[1], acc[mi][1]);
    }
  };

  issue(0, 0u);
  unsigned oiss = bstepB;   // buffer offset for next issue (uniform toggle)
  unsigned ocmp = 0u;       // buffer offset (elements) for current compute
#pragma unroll 1
  for (int t = 0; t < 31; ++t) {
    issue((t + 1) * 32, oiss);
    oiss = bstepB - oiss;
    WAIT_ASYNC_LE4();                 // tile t (oldest 4) has landed
    __syncthreads();
    compute(&sA[0][ocmp], &sB[0][ocmp]);
    ocmp = bstepE - ocmp;
    __syncthreads();                  // safe to overwrite this buf at t+2
  }
  WAIT_ASYNC_0();                     // tile 31
  __syncthreads();
  compute(&sA[0][bstepE], &sB[0][bstepE]);   // tile 31 lives in buffer 1

#pragma unroll
  for (int ni = 0; ni < 2; ++ni) {
    int n = nb + wn + ni * 16 + half16;
    float bval = bias[n];
    int h  = n >> 6;
    int hd = n & 63;
#pragma unroll
    for (int mi = 0; mi < 4; ++mi) {
      int rbase = mb + wm + mi * 16 + ((lane < 16) ? 0 : 8);
#pragma unroll
      for (int i = 0; i < 8; ++i) {
        int m  = rbase + i;
        int bb = m >> 11;       // batch
        int l  = m & 2047;      // seq pos
        float vv = acc[mi][ni][i] + bval;
        if (z == 0) {           // Q [B,H,L,HD], pre-scaled by HD^-0.5
          qo[((size_t)(bb * H_ + h) * L_ + l) * HD_ + hd] = f2bf(vv * 0.125f);
        } else if (z == 1) {    // K [B,H,L,HD]
          ko[((size_t)(bb * H_ + h) * L_ + l) * HD_ + hd] = f2bf(vv);
        } else {                // V^T [B,H,HD,L]
          vto[((size_t)(bb * H_ + h) * HD_ + hd) * L_ + l] = f2bf(vv);
        }
      }
    }
  }
}

// ---------------- flash attention: one WG per (b,h, 128 query rows) ----------------
__global__ __launch_bounds__(256) void attn_kernel(
    const __bf16* __restrict__ qb, const __bf16* __restrict__ kb,
    const __bf16* __restrict__ vtb, __bf16* __restrict__ ab) {
  __shared__ __bf16 sK[2][64 * 72];   // [key][hd], stride 144B: conflict-free
  __shared__ __bf16 sV[2][64 * 72];   // [hd][key]
  __shared__ __bf16 sP[128 * 72];     // wave-private stripes, C->A layout bounce

  const int bh = blockIdx.y;
  const int b = bh >> 4;
  const int h = bh & 15;
  const int qbase = blockIdx.x * 128;
  const int tid = threadIdx.x;
  const int lane = tid & 31;
  const int wid = tid >> 5;
  const int half16 = lane & 15;
  const int ak  = (lane < 16) ? 0 : 8;
  const int bks = (lane < 16) ? 0 : 8;

  const __bf16* Q  = qb  + (size_t)bh * L_ * HD_;
  const __bf16* K  = kb  + (size_t)bh * L_ * HD_;
  const __bf16* Vt = vtb + (size_t)bh * HD_ * L_;

  // Q fragments for this wave's 16 rows, kept in registers across the key loop
  const int qrow = qbase + wid * 16 + half16;
  v16bf qa[2];
#pragma unroll
  for (int kd = 0; kd < 2; ++kd) {
    qa[kd] = cat8(
        *reinterpret_cast<const v8bf*>(&Q[(size_t)qrow * HD_ + kd * 32 + ak]),
        *reinterpret_cast<const v8bf*>(&Q[(size_t)qrow * HD_ + kd * 32 + ak + 16]));
  }

  v8f o[4];
  float m_[8], l_[8];
#pragma unroll
  for (int i = 0; i < 4; ++i) o[i] = zero8();
#pragma unroll
  for (int i = 0; i < 8; ++i) { m_[i] = -3.0e38f; l_[i] = 0.f; }

  const int lr = tid >> 2;
  const int c8 = (tid & 3) * 8;
  __bf16* pbase = &sP[wid * 16 * 72];

  const __bf16* gk = &K[(size_t)lr * HD_ + c8];    // advance by kt*HD_
  const __bf16* gv = &Vt[(size_t)lr * L_ + c8];    // advance by kt
  const unsigned lk = (unsigned)(size_t)&sK[0][lr * 72 + c8];
  const unsigned lv = (unsigned)(size_t)&sV[0][lr * 72 + c8];
  const unsigned kstepB = 64 * 72 * 2;             // bytes between double buffers
  const unsigned kstepE = 64 * 72;                 // elements between buffers

  auto issue = [&](int kt, unsigned o2) {
    const __bf16* gkt = gk + (size_t)kt * HD_;
    const __bf16* gvt = gv + kt;
    async_ld128(lk + o2,      gkt);
    async_ld128(lk + o2 + 64, gkt + 32);           // +32 elems = +64B in LDS row
    async_ld128(lv + o2,      gvt);
    async_ld128(lv + o2 + 64, gvt + 32);
  };

  auto process = [&](const __bf16* Kb, const __bf16* Vb) {
    // S = (Q*scale) @ K^T : 16 rows x 64 keys, contract over hd (2 k-steps)
    v8f s[4];
#pragma unroll
    for (int n = 0; n < 4; ++n) {
      s[n] = zero8();
      int col = n * 16 + half16;
#pragma unroll
      for (int kd = 0; kd < 2; ++kd) {
        v16bf kf = cat8(
            *reinterpret_cast<const v8bf*>(&Kb[col * 72 + kd * 32 + bks]),
            *reinterpret_cast<const v8bf*>(&Kb[col * 72 + kd * 32 + bks + 16]));
        s[n] = wmma_bf16(qa[kd], kf, s[n]);
      }
    }

    // online softmax: row stats live entirely within 16-lane halves
    float mnew[8], alpha[8];
#pragma unroll
    for (int i = 0; i < 8; ++i) {
      float rm = fmaxf(fmaxf(s[0][i], s[1][i]), fmaxf(s[2][i], s[3][i]));
      rm = fmaxf(rm, __shfl_xor(rm, 8, 16));
      rm = fmaxf(rm, __shfl_xor(rm, 4, 16));
      rm = fmaxf(rm, __shfl_xor(rm, 2, 16));
      rm = fmaxf(rm, __shfl_xor(rm, 1, 16));
      mnew[i]  = fmaxf(m_[i], rm);
      alpha[i] = __expf(m_[i] - mnew[i]);
      m_[i] = mnew[i];
    }
#pragma unroll
    for (int n = 0; n < 4; ++n)
#pragma unroll
      for (int i = 0; i < 8; ++i)
        s[n][i] = __expf(s[n][i] - mnew[i]);
#pragma unroll
    for (int i = 0; i < 8; ++i) {
      float rs = s[0][i] + s[1][i] + s[2][i] + s[3][i];
      rs += __shfl_xor(rs, 8, 16);
      rs += __shfl_xor(rs, 4, 16);
      rs += __shfl_xor(rs, 2, 16);
      rs += __shfl_xor(rs, 1, 16);
      l_[i] = l_[i] * alpha[i] + rs;
#pragma unroll
      for (int hh = 0; hh < 4; ++hh) o[hh][i] *= alpha[i];
    }

    // bounce P through wave-private LDS stripe: C-layout -> A-layout
#pragma unroll
    for (int n = 0; n < 4; ++n)
#pragma unroll
      for (int i = 0; i < 8; ++i) {
        int r = i + ((lane < 16) ? 0 : 8);
        pbase[r * 72 + n * 16 + half16] = f2bf(s[n][i]);
      }
    asm volatile("s_wait_dscnt 0x0" ::: "memory");

    // O += P @ V : contract over 64 keys (2 k-steps), 4 hd-fragments
#pragma unroll
    for (int kd = 0; kd < 2; ++kd) {
      v16bf pf = cat8(
          *reinterpret_cast<const v8bf*>(&pbase[half16 * 72 + kd * 32 + ak]),
          *reinterpret_cast<const v8bf*>(&pbase[half16 * 72 + kd * 32 + ak + 16]));
#pragma unroll
      for (int hh = 0; hh < 4; ++hh) {
        int col = hh * 16 + half16;
        v16bf vf = cat8(
            *reinterpret_cast<const v8bf*>(&Vb[col * 72 + kd * 32 + bks]),
            *reinterpret_cast<const v8bf*>(&Vb[col * 72 + kd * 32 + bks + 16]));
        o[hh] = wmma_bf16(pf, vf, o[hh]);
      }
    }
  };

  issue(0, 0u);
  unsigned oiss = kstepB;
  unsigned ocmp = 0u;
#pragma unroll 1
  for (int t = 0; t < (L_ / 64) - 1; ++t) {
    issue((t + 1) * 64, oiss);
    oiss = kstepB - oiss;
    WAIT_ASYNC_LE4();
    __syncthreads();
    process(&sK[0][ocmp], &sV[0][ocmp]);
    ocmp = kstepE - ocmp;
    __syncthreads();
  }
  WAIT_ASYNC_0();                     // last tile (odd buffer)
  __syncthreads();
  process(&sK[0][kstepE], &sV[0][kstepE]);

  // normalize and store to attention output [B, L, D] (bf16)
#pragma unroll
  for (int hh = 0; hh < 4; ++hh) {
#pragma unroll
    for (int i = 0; i < 8; ++i) {
      int r   = qbase + wid * 16 + i + ((lane < 16) ? 0 : 8);
      int col = h * HD_ + hh * 16 + half16;
      ab[((size_t)b * L_ + r) * D_ + col] = f2bf(o[hh][i] / l_[i]);
    }
  }
}

// ---------------- output projection: out = attn @ Wo^T + bo (f32 out) ----------------
__global__ __launch_bounds__(256) void oproj_kernel(
    const __bf16* __restrict__ ah, const __bf16* __restrict__ wo,
    const float* __restrict__ bo, float* __restrict__ out) {
  __shared__ __bf16 sA[2][128 * 40];
  __shared__ __bf16 sB[2][128 * 40];

  const int mb = blockIdx.x * 128;
  const int nb = blockIdx.y * 128;
  const int tid = threadIdx.x;
  const int lane = tid & 31;
  const int wid = tid >> 5;
  const int wm = (wid & 1) * 64;
  const int wn = (wid >> 1) * 32;
  const int half16 = lane & 15;
  const int ak  = (lane < 16) ? 0 : 8;
  const int bk_ = (lane < 16) ? 0 : 8;
  const int lr = tid >> 2;
  const int lc = (tid & 3) * 8;

  v8f acc[4][2];
#pragma unroll
  for (int mi = 0; mi < 4; ++mi) { acc[mi][0] = zero8(); acc[mi][1] = zero8(); }

  const __bf16* ga0 = &ah[(size_t)(mb + lr) * D_ + lc];
  const __bf16* ga1 = &ah[(size_t)(mb + lr + 64) * D_ + lc];
  const __bf16* gb0 = &wo[(size_t)(nb + lr) * D_ + lc];
  const __bf16* gb1 = &wo[(size_t)(nb + lr + 64) * D_ + lc];
  const unsigned la0 = (unsigned)(size_t)&sA[0][lr * 40 + lc];
  const unsigned la1 = (unsigned)(size_t)&sA[0][(lr + 64) * 40 + lc];
  const unsigned lb0 = (unsigned)(size_t)&sB[0][lr * 40 + lc];
  const unsigned lb1 = (unsigned)(size_t)&sB[0][(lr + 64) * 40 + lc];
  const unsigned bstepB = 128 * 40 * 2;
  const unsigned bstepE = 128 * 40;

  auto issue = [&](int k0, unsigned o) {
    async_ld128(la0 + o, ga0 + k0);
    async_ld128(la1 + o, ga1 + k0);
    async_ld128(lb0 + o, gb0 + k0);
    async_ld128(lb1 + o, gb1 + k0);
  };

  auto compute = [&](const __bf16* A, const __bf16* B) {
    v16bf bfrag[2];
#pragma unroll
    for (int ni = 0; ni < 2; ++ni) {
      int col = wn + ni * 16 + half16;
      bfrag[ni] = cat8(*reinterpret_cast<const v8bf*>(&B[col * 40 + bk_]),
                       *reinterpret_cast<const v8bf*>(&B[col * 40 + bk_ + 16]));
    }
#pragma unroll
    for (int mi = 0; mi < 4; ++mi) {
      int row = wm + mi * 16 + half16;
      v16bf afrag = cat8(*reinterpret_cast<const v8bf*>(&A[row * 40 + ak]),
                         *reinterpret_cast<const v8bf*>(&A[row * 40 + ak + 16]));
      acc[mi][0] = wmma_bf16(afrag, bfrag[0], acc[mi][0]);
      acc[mi][1] = wmma_bf16(afrag, bfrag[1], acc[mi][1]);
    }
  };

  issue(0, 0u);
  unsigned oiss = bstepB;
  unsigned ocmp = 0u;
#pragma unroll 1
  for (int t = 0; t < 31; ++t) {
    issue((t + 1) * 32, oiss);
    oiss = bstepB - oiss;
    WAIT_ASYNC_LE4();
    __syncthreads();
    compute(&sA[0][ocmp], &sB[0][ocmp]);
    ocmp = bstepE - ocmp;
    __syncthreads();
  }
  WAIT_ASYNC_0();
  __syncthreads();
  compute(&sA[0][bstepE], &sB[0][bstepE]);

#pragma unroll
  for (int ni = 0; ni < 2; ++ni) {
    int n = nb + wn + ni * 16 + half16;
    float bval = bo[n];
#pragma unroll
    for (int mi = 0; mi < 4; ++mi) {
      int rbase = mb + wm + mi * 16 + ((lane < 16) ? 0 : 8);
#pragma unroll
      for (int i = 0; i < 8; ++i) {
        int m = rbase + i;
        out[(size_t)m * D_ + n] = acc[mi][ni][i] + bval;
      }
    }
  }
}

extern "C" void kernel_launch(void* const* d_in, const int* in_sizes, int n_in,
                              void* d_out, int out_size, void* d_ws, size_t ws_size,
                              hipStream_t stream) {
  (void)in_sizes; (void)n_in; (void)out_size; (void)ws_size;
  const float* x  = (const float*)d_in[0];
  const float* Wq = (const float*)d_in[1];
  const float* bq = (const float*)d_in[2];
  const float* Wk = (const float*)d_in[3];
  const float* bk = (const float*)d_in[4];
  const float* Wv = (const float*)d_in[5];
  const float* bv = (const float*)d_in[6];
  const float* Wo = (const float*)d_in[7];
  const float* bo = (const float*)d_in[8];
  float* out = (float*)d_out;

  char* ws = (char*)d_ws;
  const size_t MB = (size_t)1 << 20;
  __bf16* xh  = (__bf16*)(ws + 0 * MB);   // 8 MB (reused as attn output after QKV)
  __bf16* wqh = (__bf16*)(ws + 8 * MB);   // 2 MB each
  __bf16* wkh = (__bf16*)(ws + 10 * MB);
  __bf16* wvh = (__bf16*)(ws + 12 * MB);
  __bf16* woh = (__bf16*)(ws + 14 * MB);
  __bf16* qh  = (__bf16*)(ws + 16 * MB);  // [B,H,L,HD] 8 MB
  __bf16* kh  = (__bf16*)(ws + 24 * MB);  // [B,H,L,HD] 8 MB
  __bf16* vth = (__bf16*)(ws + 32 * MB);  // [B,H,HD,L] 8 MB
  __bf16* ah  = xh;                       // attention output [B,L,D] bf16

  cvt_kernel<<<4096, 256, 0, stream>>>(x,  xh,  (2 * L_ * D_) / 4);
  cvt_kernel<<<1024, 256, 0, stream>>>(Wq, wqh, (D_ * D_) / 4);
  cvt_kernel<<<1024, 256, 0, stream>>>(Wk, wkh, (D_ * D_) / 4);
  cvt_kernel<<<1024, 256, 0, stream>>>(Wv, wvh, (D_ * D_) / 4);
  cvt_kernel<<<1024, 256, 0, stream>>>(Wo, woh, (D_ * D_) / 4);

  qkv_kernel<<<dim3(32, 8, 3), 256, 0, stream>>>(xh, wqh, wkh, wvh,
                                                 bq, bk, bv, qh, kh, vth);
  attn_kernel<<<dim3(16, 32), 256, 0, stream>>>(qh, kh, vth, ah);
  oproj_kernel<<<dim3(32, 8), 256, 0, stream>>>(ah, woh, bo, out);
}
